// TopKRouter_37589553774751
// MI455X (gfx1250) — matrix operand order repaired and checked by reading, not compiled, not run on hardware
//
#include <hip/hip_runtime.h>
#include <math.h>

typedef __attribute__((ext_vector_type(16))) _Float16     v16h;
typedef __attribute__((ext_vector_type(8)))  float        v8f;
typedef __attribute__((ext_vector_type(4)))  unsigned int v4u;

#define T_TOKENS 32768
#define D_MODEL  4096
#define N_EXP    64
#define KCHUNK   256
#define WSTRIDE  (KCHUNK + 8)   // halves; 528 B row stride (== 16 mod 256) rotates banks
#define QPR      (KCHUNK / 8)   // 16 B chunks per expert row = 32
#define SC_STRIDE 68            // floats; 272 B row stride rotates banks for top-k scan
#define WAVES_PER_BLOCK 8
#define TOKENS_PER_WAVE 16
#define TOKENS_PER_BLOCK (WAVES_PER_BLOCK * TOKENS_PER_WAVE)
#define NTHREADS (WAVES_PER_BLOCK * 32)

union FragU { v16h v; v4u q[2]; };

// CDNA5 async memory->LDS copy (GVS mode), tracked by ASYNCcnt.
// LDS dest offset comes from the vdst VGPR; global address is SGPR base plus VGPR offset
__device__ __forceinline__ void async_copy_b128(unsigned lds_off, unsigned gvoff,
                                                const void* sbase) {
    asm volatile("global_load_async_to_lds_b128 %0, %1, %2 offset:0"
                 :
                 : "v"(lds_off), "v"(gvoff), "s"(sbase)
                 : "memory");
}

__device__ __forceinline__ void wait_async_zero() {
    asm volatile("s_wait_asynccnt 0x0" ::: "memory");
}

__global__ __launch_bounds__(NTHREADS)
void topk_router_kernel(const _Float16* __restrict__ x,
                        const _Float16* __restrict__ W,
                        int*   __restrict__ idx_out,
                        float* __restrict__ w_out) {
    __shared__ _Float16 w_lds[2][N_EXP * WSTRIDE];                        // 2 x 33 KB
    __shared__ float    sc[WAVES_PER_BLOCK][TOKENS_PER_WAVE * SC_STRIDE]; // ~34 KB

    const int tid  = threadIdx.x;
    const int lane = tid & 31;
    const int wave = tid >> 5;
    const int hi   = lane >> 4;      // 0: lanes 0-15, 1: lanes 16-31
    const int l15  = lane & 15;

    const int tok0 = blockIdx.x * TOKENS_PER_BLOCK + wave * TOKENS_PER_WAVE;

    // A-matrix: this lane always feeds token row (tok0 + l15)
    const _Float16* xrow = x + (size_t)(tok0 + l15) * D_MODEL;

    // wave-relative LDS byte offsets of the two W buffers (generic ptr low 32 bits)
    const unsigned wbase0 = (unsigned)(uintptr_t)(&w_lds[0][0]);
    const unsigned wbase1 = (unsigned)(uintptr_t)(&w_lds[1][0]);

    // staging decomposition: flat index i = 256*j + tid -> e = wave + 8*j, c = lane
    const unsigned st_lds = (unsigned)(wave * (WSTRIDE * 2) + lane * 16);
    const unsigned st_gbl = (unsigned)(wave * (D_MODEL * 2) + lane * 16);

    v8f acc0 = {}, acc1 = {}, acc2 = {}, acc3 = {};

    // ---- prologue: async-stage chunk 0 into buffer 0 ----
    #pragma unroll
    for (int j = 0; j < 8; ++j) {
        async_copy_b128(wbase0 + st_lds + (unsigned)(j * 8 * WSTRIDE * 2),
                        st_gbl + (unsigned)(j * 8 * D_MODEL * 2), W);
    }
    wait_async_zero();
    __syncthreads();

    int buf = 0;
    for (int kc = 0; kc < D_MODEL; kc += KCHUNK) {
        // ---- async-stage next chunk into the other buffer (overlaps compute) ----
        if (kc + KCHUNK < D_MODEL) {
            const unsigned nbase = (buf == 0) ? wbase1 : wbase0;
            const unsigned gchunk = st_gbl + (unsigned)((kc + KCHUNK) * 2);
            #pragma unroll
            for (int j = 0; j < 8; ++j) {
                async_copy_b128(nbase + st_lds + (unsigned)(j * 8 * WSTRIDE * 2),
                                gchunk + (unsigned)(j * 8 * D_MODEL * 2), W);
            }
        }

        // ---- WMMA over this K chunk: 8 steps of K=32, 4 expert tiles each ----
        const _Float16* wb = &w_lds[buf][0];
        #pragma unroll 4
        for (int kk = 0; kk < KCHUNK; kk += 32) {
            // A fragment (16x32 f16): lanes 0-15 K=0..15, lanes 16-31 K=16..31 (contiguous)
            // Non-temporal: x is streamed exactly once; don't evict W from L2.
            FragU a;
            const _Float16* ap = xrow + kc + kk + hi * 16;
            a.q[0] = __builtin_nontemporal_load((const v4u*)(ap));
            a.q[1] = __builtin_nontemporal_load((const v4u*)(ap + 8));

            // B fragments (32x16 f16): col = e0 + l15, K = kk + hi*16 + [0..15] contiguous
            const int kb = kk + hi * 16;
            FragU b0, b1, b2, b3;
            const _Float16* bp0 = wb + ( 0 + l15) * WSTRIDE + kb;
            const _Float16* bp1 = wb + (16 + l15) * WSTRIDE + kb;
            const _Float16* bp2 = wb + (32 + l15) * WSTRIDE + kb;
            const _Float16* bp3 = wb + (48 + l15) * WSTRIDE + kb;
            b0.q[0] = *(const v4u*)(bp0);  b0.q[1] = *(const v4u*)(bp0 + 8);
            b1.q[0] = *(const v4u*)(bp1);  b1.q[1] = *(const v4u*)(bp1 + 8);
            b2.q[0] = *(const v4u*)(bp2);  b2.q[1] = *(const v4u*)(bp2 + 8);
            b3.q[0] = *(const v4u*)(bp3);  b3.q[1] = *(const v4u*)(bp3 + 8);

            acc0 = __builtin_amdgcn_wmma_f32_16x16x32_f16(false, a.v, false, b0.v,
                                                          (short)0, acc0, false, false);
            acc1 = __builtin_amdgcn_wmma_f32_16x16x32_f16(false, a.v, false, b1.v,
                                                          (short)0, acc1, false, false);
            acc2 = __builtin_amdgcn_wmma_f32_16x16x32_f16(false, a.v, false, b2.v,
                                                          (short)0, acc2, false, false);
            acc3 = __builtin_amdgcn_wmma_f32_16x16x32_f16(false, a.v, false, b3.v,
                                                          (short)0, acc3, false, false);
        }

        // next buffer fully written + everyone done reading current before overwrite
        wait_async_zero();
        __syncthreads();
        buf ^= 1;
    }

    // ---- dump scores to LDS: D layout is m = r + hi*8, n = l15 (per 16-wide tile) ----
    float* scw = sc[wave];
    #pragma unroll
    for (int r = 0; r < 8; ++r) {
        const int m = r + hi * 8;
        scw[m * SC_STRIDE +  0 + l15] = acc0[r];
        scw[m * SC_STRIDE + 16 + l15] = acc1[r];
        scw[m * SC_STRIDE + 32 + l15] = acc2[r];
        scw[m * SC_STRIDE + 48 + l15] = acc3[r];
    }
    __syncthreads();

    // ---- top-2 + softmax, one token per lane (lanes 0-15) ----
    if (lane < TOKENS_PER_WAVE) {
        const float* row = &scw[lane * SC_STRIDE];
        float v1 = -__builtin_inff(), v2 = -__builtin_inff();
        int   i1 = 0, i2 = 0;
        #pragma unroll 8
        for (int e = 0; e < N_EXP; ++e) {
            const float v = row[e];
            if (v > v1)      { v2 = v1; i2 = i1; v1 = v; i1 = e; }
            else if (v > v2) { v2 = v;  i2 = e; }
        }
        const float ex    = __expf(v2 - v1);
        const float denom = 1.0f + ex;
        const int token = tok0 + lane;
        __builtin_nontemporal_store(i1, &idx_out[token * 2 + 0]);
        __builtin_nontemporal_store(i2, &idx_out[token * 2 + 1]);
        __builtin_nontemporal_store(1.0f / denom, &w_out[token * 2 + 0]);
        __builtin_nontemporal_store(ex / denom,   &w_out[token * 2 + 1]);
    }
}

extern "C" void kernel_launch(void* const* d_in, const int* in_sizes, int n_in,
                              void* d_out, int out_size, void* d_ws, size_t ws_size,
                              hipStream_t stream) {
    (void)in_sizes; (void)n_in; (void)out_size; (void)d_ws; (void)ws_size;
    const _Float16* x = (const _Float16*)d_in[0];
    const _Float16* W = (const _Float16*)d_in[1];
    // d_in[2] is k == 2 (hardcoded top-2 path)

    int*   idx_out = (int*)d_out;                              // [T,2] int32 bits
    float* w_out   = (float*)d_out + (size_t)T_TOKENS * 2;     // [T,2] fp32

    dim3 grid(T_TOKENS / TOKENS_PER_BLOCK);   // 256 blocks
    dim3 block(NTHREADS);                     // 256 threads = 8 wave32
    topk_router_kernel<<<grid, block, 0, stream>>>(x, W, idx_out, w_out);
}
